// MultiHeadAttention_75136157876548
// MI455X (gfx1250) — matrix operand order repaired and checked by reading, not compiled, not run on hardware
//
#include <hip/hip_runtime.h>
#include <hip/hip_bf16.h>

// CDNA5 / gfx1250 wave32 implementation of MHA + output projection.
//   - v_wmma_f32_16x16x32_f16 for QK^T, P@V, and O@W^T (f32 accumulate)
//   - Tensor Data Mover (tensor_load_to_lds) stages Q/K/V tiles into LDS,
//     double-buffered under TENSORcnt, with LDS padding for bank-conflict-free
//     fragment gathers.
// d_in: Q[f32 B*SQ*DM], K[f32 B*SK*DM], V[f32 B*SK*DM], mask[bool B*SQ*SK], W[f32 DM*DM]
// d_out: out[f32 B*SQ*DM] ++ attn[f32 B*H*SQ*SK]

typedef __attribute__((ext_vector_type(16))) _Float16     v16h;
typedef __attribute__((ext_vector_type(8)))  float        v8f;
typedef __attribute__((ext_vector_type(4)))  unsigned int u32x4;
typedef __attribute__((ext_vector_type(8)))  int          i32x8;
typedef __attribute__((ext_vector_type(4)))  int          i32x4;

#define B_  2
#define SQ_ 2048
#define SK_ 2048
#define H_  16
#define DK_ 64
#define DV_ 64
#define DM_ 1024
#define NEGV (-1e9f)

// LDS geometry (floats). TDM pad: +3 dwords every 64 dwords -> row stride 67.
#define SSTR  2068                 // score row stride (16B aligned, conflict-free)
#define KSTR  67                   // padded tile row stride
#define QTILE (16 * KSTR)          // 1072
#define KTILE (16 * KSTR)          // 1072
#define VTILE (32 * KSTR)          // 2144
#define OFF_SRED (16 * SSTR)                   // 33088
#define OFF_QF   (OFF_SRED + 8 * 256)          // 35136
#define OFF_K    (OFF_QF + QTILE)              // 36208
#define OFF_V    (OFF_K + 8 * 2 * KTILE)       // 53360
#define SMEM_FLOATS (OFF_V + 4 * VTILE)        // 61936 -> 247744 bytes

// A-fragment K offset pattern for v_wmma_*_16x16x32_f16 (ISA 7.12.2):
// lanes 0-15 hold K = {0..7, 16..23}; lanes 16-31 hold K = {8..15, 24..31}.
__device__ __forceinline__ int a_koff(int v, int kb) {
    return kb + (v < 4 ? 2 * v : 16 + 2 * (v - 4));
}

// Low 32 bits of a generic pointer to an LDS object = LDS byte offset
// (LDS aperture lives in the high bits), which is what D#.lds_addr wants.
__device__ __forceinline__ unsigned lds_off(const void* p) {
    return (unsigned)(size_t)p;
}

// TDM: async DMA of a 2D tile (rows x w_elems f32, row stride stride_elems)
// from global memory into LDS. pad=true inserts 3 dwords after every 64
// dwords in LDS (row stride 67 for 64-wide tiles). Tracked by TENSORcnt.
__device__ __forceinline__ void tdm_load_2d(unsigned lds_addr, const void* gptr,
                                            unsigned w_elems, unsigned rows,
                                            unsigned stride_elems, bool pad) {
    const unsigned long long ga = (unsigned long long)(size_t)gptr;
    u32x4 g0;
    g0[0] = 1u;                                                 // count=1, user mode
    g0[1] = lds_addr;                                           // D#.lds_addr
    g0[2] = (unsigned)ga;                                       // global_addr[31:0]
    g0[3] = (unsigned)((ga >> 32) & 0x1FFFFFFu) | (2u << 30);   // addr[56:32] | type=2
    i32x8 g1;
    // data_size=2 (4B) [17:16]; pad_enable [20]; pad_interval=5 (64 dw) [24:22];
    // pad_amount=2 (3 dwords) [31:25]
    g1[0] = (int)(0x20000u | (pad ? ((1u << 20) | (5u << 22) | (2u << 25)) : 0u));
    g1[1] = (int)(w_elems << 16);                    // tensor_dim0[15:0]
    g1[2] = (int)((w_elems >> 16) | (rows << 16));   // tensor_dim0[31:16] | tensor_dim1[15:0]
    g1[3] = (int)((rows >> 16) | (w_elems << 16));   // tensor_dim1[31:16] | tile_dim0
    g1[4] = (int)rows;                               // tile_dim1 | tile_dim2=0
    g1[5] = (int)stride_elems;                       // tensor_dim0_stride[31:0]
    g1[6] = 0;                                       // stride hi | dim1_stride lo (2D: unused)
    g1[7] = 0;
    i32x4 z4 = {0, 0, 0, 0};
#if __clang_major__ >= 23
    i32x8 z8 = {0, 0, 0, 0, 0, 0, 0, 0};
    __builtin_amdgcn_tensor_load_to_lds(g0, g1, z4, z4, z8, 0);
#else
    __builtin_amdgcn_tensor_load_to_lds(g0, g1, z4, z4, 0);
#endif
}

// ---------------------------------------------------------------------------
// Kernel 1: fused scores -> mask -> softmax -> attn write -> P @ V
// Block = 256 threads (8 waves). One block per (b, h, 16-row query tile).
// ---------------------------------------------------------------------------
__global__ __launch_bounds__(256) void attn_fused_kernel(
    const float* __restrict__ Q, const float* __restrict__ K,
    const float* __restrict__ V, const unsigned char* __restrict__ mask,
    float* __restrict__ attn, _Float16* __restrict__ Oh)
{
    extern __shared__ float smem[];
    float* sS   = smem;                // [16][SSTR] scores / probs
    float* sRed = smem + OFF_SRED;     // [8][256] split-K partials
    float* sQf  = smem + OFF_QF;       // TDM-staged Q tile (padded f32)
    float* sK   = smem + OFF_K;        // 8 waves x 2 bufs x KTILE
    float* sV   = smem + OFF_V;        // 2 streams x 2 bufs x VTILE
    __shared__ _Float16 sQh[16 * DK_];

    const int tid  = threadIdx.x;
    const int w    = tid >> 5;
    const int lane = tid & 31;

    const int bh    = blockIdx.x >> 7;     // / (SQ_/16)
    const int qTile = blockIdx.x & 127;
    const int b     = bh >> 4;             // / H_
    const int h     = bh & 15;
    const int qBase = qTile * 16;

    const float* Qp = Q + ((size_t)b * SQ_ + qBase) * DM_ + h * DK_;
    const float* Kp = K + (size_t)b * SK_ * DM_ + h * DK_;
    const float* Vp = V + (size_t)b * SK_ * DM_ + h * DV_;
    const unsigned char* Mp = mask + ((size_t)b * SQ_ + qBase) * SK_;
    float*     Ap = attn + (((size_t)b * H_ + h) * SQ_ + qBase) * SK_;
    _Float16*  Op = Oh + ((size_t)b * SQ_ + qBase) * DM_ + h * DV_;

    // ---- TDM-stage the 16x64 Q tile, then convert to f16 in LDS ----
    if (w == 0) {
        tdm_load_2d(lds_off(sQf), Qp, DK_, 16, DM_, true);
        __builtin_amdgcn_s_wait_tensorcnt(0);
    }
    __syncthreads();
    for (int i = tid; i < 16 * DK_; i += 256)
        sQh[i] = (_Float16)sQf[(i >> 6) * KSTR + (i & 63)];
    __syncthreads();

    const int m  = lane & 15;
    const int hi = lane >> 4;
    const int kb = hi * 8;
    const int kl = hi * 16;

    // A fragments (Q tile) for d = 0..31 and 32..63, reused across all n-tiles.
    v16h a0, a1;
    #pragma unroll
    for (int v = 0; v < 8; ++v) {
        const int kk = a_koff(v, kb);
        a0[2 * v]     = sQh[m * DK_ + kk];
        a0[2 * v + 1] = sQh[m * DK_ + kk + 1];
        a1[2 * v]     = sQh[m * DK_ + kk + 32];
        a1[2 * v + 1] = sQh[m * DK_ + kk + 33];
    }

    // ---- phase 1: scores. Each wave owns a 16x256 column slice; its K tiles
    // are TDM double-buffered in its private LDS region (no barriers needed,
    // TENSORcnt is per-wave). ----
    float* myK = sK + w * (2 * KTILE);
    tdm_load_2d(lds_off(myK), Kp + (size_t)(w * 256) * DM_, DK_, 16, DM_, true);
    for (int t = 0; t < 16; ++t) {
        if (t < 15) {
            tdm_load_2d(lds_off(myK + ((t + 1) & 1) * KTILE),
                        Kp + (size_t)(w * 256 + (t + 1) * 16) * DM_, DK_, 16, DM_, true);
            __builtin_amdgcn_s_wait_tensorcnt(1);   // tile t landed
        } else {
            __builtin_amdgcn_s_wait_tensorcnt(0);
        }
        const int   col0 = w * 256 + t * 16;
        const float* krow = myK + (t & 1) * KTILE + m * KSTR;   // K row col0+m (padded)
        v16h b0, b1;
        #pragma unroll
        for (int v = 0; v < 8; ++v) {
            const int d = a_koff(v, kb);
            b0[2 * v]     = (_Float16)krow[d];
            b0[2 * v + 1] = (_Float16)krow[d + 1];
            b1[2 * v]     = (_Float16)krow[d + 32];
            b1[2 * v + 1] = (_Float16)krow[d + 33];
        }
        v8f c = {};
        c = __builtin_amdgcn_wmma_f32_16x16x32_f16(false, a0, false, b0, (short)0, c, false, false);
        c = __builtin_amdgcn_wmma_f32_16x16x32_f16(false, a1, false, b1, (short)0, c, false, false);
        #pragma unroll
        for (int r = 0; r < 8; ++r) {
            const int mm  = r + hi * 8;            // C/D: M = vgpr + 8*(lane>=16)
            const int col = col0 + m;              // N = lane&15
            float s = c[r] * 0.125f;               // 1/sqrt(64)
            if (Mp[(size_t)mm * SK_ + col]) s = NEGV;
            sS[mm * SSTR + col] = s;
        }
    }
    __syncthreads();

    // ---- softmax: 16 threads per row, interleaved float4 segments
    // (conflict-free LDS, 256B-contiguous global stores per 16-lane group) ----
    {
        const int row = tid >> 4;
        const int seg = tid & 15;
        float* prow = sS + row * SSTR;
        float mx = -3.0e38f;
        for (int j = 0; j < 32; ++j) {
            const float4 p = *(const float4*)(prow + seg * 4 + j * 64);
            mx = fmaxf(mx, fmaxf(fmaxf(p.x, p.y), fmaxf(p.z, p.w)));
        }
        #pragma unroll
        for (int off = 8; off >= 1; off >>= 1) mx = fmaxf(mx, __shfl_xor(mx, off, 16));
        float sum = 0.f;
        for (int j = 0; j < 32; ++j) {
            float4 p = *(float4*)(prow + seg * 4 + j * 64);
            p.x = __expf(p.x - mx); p.y = __expf(p.y - mx);
            p.z = __expf(p.z - mx); p.w = __expf(p.w - mx);
            sum += (p.x + p.y) + (p.z + p.w);
            *(float4*)(prow + seg * 4 + j * 64) = p;
        }
        #pragma unroll
        for (int off = 8; off >= 1; off >>= 1) sum += __shfl_xor(sum, off, 16);
        const float inv = 1.f / sum;
        float* arow = Ap + (size_t)row * SK_;
        for (int j = 0; j < 32; ++j) {
            float4 p = *(float4*)(prow + seg * 4 + j * 64);
            p.x *= inv; p.y *= inv; p.z *= inv; p.w *= inv;
            *(float4*)(prow + seg * 4 + j * 64) = p;   // keep P for phase 2
            *(float4*)(arow + seg * 4 + j * 64) = p;   // attn output
        }
    }
    __syncthreads();

    // ---- phase 2: O = P @ V. Wave w: n-tile (w&3), K-half (w>>2).
    // V chunks (32 rows x 64) TDM-staged by waves 0 (rows 0..1023) and
    // 4 (rows 1024..2047), double-buffered per stream. ----
    const int nt = w & 3;
    const int kh = w >> 2;
    if (w == 0) tdm_load_2d(lds_off(sV), Vp, DV_, 32, DM_, true);
    if (w == 4) tdm_load_2d(lds_off(sV + 2 * VTILE), Vp + (size_t)1024 * DM_, DV_, 32, DM_, true);
    v8f oc = {};
    for (int i = 0; i < 32; ++i) {
        if (i < 31) {
            if (w == 0)
                tdm_load_2d(lds_off(sV + ((i + 1) & 1) * VTILE),
                            Vp + (size_t)((i + 1) * 32) * DM_, DV_, 32, DM_, true);
            if (w == 4)
                tdm_load_2d(lds_off(sV + (2 + ((i + 1) & 1)) * VTILE),
                            Vp + (size_t)(1024 + (i + 1) * 32) * DM_, DV_, 32, DM_, true);
            if ((w & 3) == 0) __builtin_amdgcn_s_wait_tensorcnt(1);
        } else {
            if ((w & 3) == 0) __builtin_amdgcn_s_wait_tensorcnt(0);
        }
        __syncthreads();   // chunk i visible to all waves
        const float* vtile = sV + (kh * 2 + (i & 1)) * VTILE;
        const int base = (kh * 32 + i) * 32;
        v16h pa, vb;
        #pragma unroll
        for (int v = 0; v < 8; ++v) {
            const int kk = base + a_koff(v, kb);
            pa[2 * v]     = (_Float16)sS[m * SSTR + kk];
            pa[2 * v + 1] = (_Float16)sS[m * SSTR + kk + 1];
        }
        const int cc = nt * 16 + m;
        #pragma unroll
        for (int v = 0; v < 8; ++v) {
            const int r = a_koff(v, kb);
            vb[2 * v]     = (_Float16)vtile[r * KSTR + cc];
            vb[2 * v + 1] = (_Float16)vtile[(r + 1) * KSTR + cc];
        }
        oc = __builtin_amdgcn_wmma_f32_16x16x32_f16(false, pa, false, vb, (short)0, oc, false, false);
        __syncthreads();   // all reads done before buffer is re-filled
    }
    // split-K reduce: wave pairs (w, w+4) share an n-tile.
    #pragma unroll
    for (int r = 0; r < 8; ++r)
        sRed[w * 256 + (r + hi * 8) * 16 + m] = oc[r];
    __syncthreads();
    if (w < 4) {
        #pragma unroll
        for (int r = 0; r < 8; ++r) {
            const int mm = r + hi * 8;
            const float tot = oc[r] + sRed[(w + 4) * 256 + mm * 16 + m];
            Op[(size_t)mm * DM_ + nt * 16 + m] = (_Float16)tot;
        }
    }
}

// ---------------------------------------------------------------------------
// Kernel 2: out = Oh @ W^T  (Oh: [B*SQ][DM] f16 in workspace, W: [DM][DM] f32)
// 8 waves per block, one 16x16 output tile per wave, K = 1024 (32 WMMA steps).
// ---------------------------------------------------------------------------
__global__ __launch_bounds__(256) void proj_gemm_kernel(
    const _Float16* __restrict__ Oh, const float* __restrict__ Wm,
    float* __restrict__ out)
{
    const int tid  = threadIdx.x;
    const int w    = tid >> 5;
    const int lane = tid & 31;
    const int tile = blockIdx.x * 8 + w;
    const int rowTile = tile >> 6;        // / (DM_/16)
    const int colTile = tile & 63;
    const int m  = lane & 15;
    const int hi = lane >> 4;
    const int kb = hi * 8;
    const int kl = hi * 16;

    const _Float16* arow = Oh + (size_t)(rowTile * 16 + m) * DM_;
    const float*    wrow = Wm + (size_t)(colTile * 16 + m) * DM_;  // B(k,n)=W[n][k]

    v8f c = {};
    for (int ch = 0; ch < 32; ++ch) {
        const int base = ch * 32;
        if (ch + 1 < 32)
            __builtin_prefetch(wrow + base + 32, 0, 1);   // global_prefetch_b8
        v16h a, bb;
        #pragma unroll
        for (int v = 0; v < 8; ++v) {
            const int kk = base + a_koff(v, kb);
            a[2 * v]     = arow[kk];
            a[2 * v + 1] = arow[kk + 1];
        }
        #pragma unroll
        for (int v = 0; v < 8; ++v) {
            const int k = base + a_koff(v, kb);
            bb[2 * v]     = (_Float16)wrow[k];
            bb[2 * v + 1] = (_Float16)wrow[k + 1];
        }
        c = __builtin_amdgcn_wmma_f32_16x16x32_f16(false, a, false, bb, (short)0, c, false, false);
    }
    #pragma unroll
    for (int r = 0; r < 8; ++r)
        out[(size_t)(rowTile * 16 + r + hi * 8) * DM_ + colTile * 16 + m] = c[r];
}

extern "C" void kernel_launch(void* const* d_in, const int* in_sizes, int n_in,
                              void* d_out, int out_size, void* d_ws, size_t ws_size,
                              hipStream_t stream) {
    (void)in_sizes; (void)n_in; (void)out_size; (void)ws_size;
    const float* Q = (const float*)d_in[0];
    const float* K = (const float*)d_in[1];
    const float* V = (const float*)d_in[2];
    const unsigned char* mask = (const unsigned char*)d_in[3];  // jnp.bool_ = 1 byte
    const float* W = (const float*)d_in[4];

    float* out  = (float*)d_out;
    float* attn = out + (size_t)B_ * SQ_ * DM_;
    _Float16* Oh = (_Float16*)d_ws;   // [B*SQ][DM] f16 head-merged context (8 MB)

    const size_t smem = (size_t)SMEM_FLOATS * sizeof(float);  // ~242 KB (of 320 KB/WGP)
    hipFuncSetAttribute(reinterpret_cast<const void*>(attn_fused_kernel),
                        hipFuncAttributeMaxDynamicSharedMemorySize, (int)smem);

    attn_fused_kernel<<<dim3(B_ * H_ * (SQ_ / 16)), dim3(256), smem, stream>>>(
        Q, K, V, mask, attn, Oh);
    proj_gemm_kernel<<<dim3((B_ * SQ_ / 16) * (DM_ / 16) / 8), dim3(256), 0, stream>>>(
        Oh, W, out);
}
